// Conv2DFromScratch_47966194762151
// MI455X (gfx1250) — matrix operand order repaired and checked by reading, not compiled, not run on hardware
//
#include <hip/hip_runtime.h>

typedef _Float16 v16h __attribute__((ext_vector_type(16)));
typedef _Float16 v8h  __attribute__((ext_vector_type(8)));
typedef float    v8f  __attribute__((ext_vector_type(8)));
typedef float    v4f  __attribute__((ext_vector_type(4)));

#define IN_CH  128
#define OUT_CH 256
#define HW     56
#define PIX    (HW * HW)        // 3136
#define KTOT   (IN_CH * 9)      // 1152
#define BM     64
#define BN     256              // all output channels in one block
#define BK     32
#define KPAD   40               // padded LDS row (halves); 80B, 16B-aligned
#define NKT    (KTOT / BK)      // 36

// ---------------------------------------------------------------------------
// Prepass: weight OIHW fp32 -> Bt[o][k'] f16 with k' = (kh*3+kw)*128 + c.
// ---------------------------------------------------------------------------
__global__ __launch_bounds__(256)
void reorder_weights_f16(const float* __restrict__ w, _Float16* __restrict__ bt) {
    int idx = blockIdx.x * 256 + threadIdx.x;   // 0 .. 294911
    int o  = idx / KTOT;
    int kp = idx - o * KTOT;
    int s  = kp >> 7;          // kh*3+kw
    int c  = kp & 127;
    bt[idx] = (_Float16)w[o * KTOT + c * 9 + s];
}

// ---------------------------------------------------------------------------
// Implicit-GEMM conv via WMMA f16 (f32 accumulate), double-buffered LDS.
// B tile moved global->LDS with GLOBAL_LOAD_ASYNC_TO_LDS_B128 (ASYNCcnt),
// A tile gathered branchlessly through registers (needs f32->f16 + mask).
// Grid: (1568, 1). Block: 256 threads = 8 wave32 in 2(M) x 4(N) grid,
// each wave computes a 32x64 tile: 8 v_wmma_f32_16x16x32_f16 per k-step.
// ---------------------------------------------------------------------------
__global__ __launch_bounds__(256)
void conv3x3_wmma_f16(const float* __restrict__ x,
                      const _Float16* __restrict__ bt,
                      const float* __restrict__ bias,
                      float* __restrict__ out) {
    __shared__ __align__(16) _Float16 As[2][BM * KPAD];   // 2 x  5 KB
    __shared__ __align__(16) _Float16 Bs[2][BN * KPAD];   // 2 x 20 KB

    const int t      = threadIdx.x;
    const int lane   = t & 31;
    const int wv     = t >> 5;       // 0..7
    const int waveM  = wv & 1;       // 32-row sub-tile
    const int waveN  = wv >> 1;      // 0..3 -> 64-col sub-tile
    const int r      = lane & 15;
    const int hiHalf = lane >> 4;    // 0 or 1

    const int mBase = blockIdx.x * BM;
    const int nImg  = mBase / PIX;   // 64 | 3136 -> block stays in one image
    const int pBase = mBase - nImg * PIX;

    // A-loader: thread covers (m_local = t&63, 8 channels at group t>>6)
    const int a_m = t & 63;
    const int a_g = t >> 6;          // 0..3 -> channel offset a_g*8
    const int p   = pBase + a_m;
    const int oh  = p / HW;
    const int ow  = p - oh * HW;
    const float* xA = x + (size_t)nImg * (IN_CH * PIX) + (size_t)(a_g * 8) * PIX;

    // B-loader: one weight row per thread (64 contiguous bytes per k-tile)
    const _Float16* btRow = bt + (size_t)t * KTOT;

    v8f acc[2][4] = {};

    // register staging (A only)
    float areg[8];
    float amask;

    auto fetchA = [&](int kt) {
        const int s    = kt >> 2;            // kh*3+kw
        const int cBlk = (kt & 3) * 32;
        const int kh   = s / 3;
        const int kw   = s - kh * 3;
        const int ih   = oh + kh - 1;
        const int iw   = ow + kw - 1;
        const bool valid = ((unsigned)ih < (unsigned)HW) &
                           ((unsigned)iw < (unsigned)HW);
        amask = valid ? 1.0f : 0.0f;
        // Branchless: always load from a legal address, zero afterwards.
        const int off = valid ? (ih * HW + iw) : 0;
        const float* src = xA + (size_t)cBlk * PIX + off;
#pragma unroll
        for (int j = 0; j < 8; ++j)
            areg[j] = src[(size_t)j * PIX];
    };

    auto stageA = [&](int buf) {
        v8h av;
#pragma unroll
        for (int j = 0; j < 8; ++j)
            av[j] = (_Float16)(areg[j] * amask);
        *(v8h*)(&As[buf][a_m * KPAD + a_g * 8]) = av;
    };

    // Async DMA: 64B of weight row -> LDS, no VGPR data movement.
    // IOFFSET is added to both the LDS and the global address (ISA 10.7).
    auto stageB_async = [&](int buf, int kt) {
        const unsigned ldsAddr =
            (unsigned)(uintptr_t)(&Bs[buf][t * KPAD]);   // low 32b = LDS offset
        const unsigned long long gaddr =
            (unsigned long long)(uintptr_t)(btRow + kt * BK);
        asm volatile(
            "global_load_async_to_lds_b128 %0, %1, off\n\t"
            "global_load_async_to_lds_b128 %0, %1, off offset:16\n\t"
            "global_load_async_to_lds_b128 %0, %1, off offset:32\n\t"
            "global_load_async_to_lds_b128 %0, %1, off offset:48"
            :: "v"(ldsAddr), "v"(gaddr) : "memory");
    };

    auto compute = [&](int buf) {
        v16h a[2];
#pragma unroll
        for (int i = 0; i < 2; ++i) {
            const _Float16* ap =
                &As[buf][(waveM * 32 + i * 16 + r) * KPAD + hiHalf * 8];
            v8h lo = *(const v8h*)ap;
            v8h hi = *(const v8h*)(ap + 16);
#pragma unroll
            for (int j = 0; j < 8; ++j) { a[i][j] = lo[j]; a[i][j + 8] = hi[j]; }
        }
#pragma unroll
        for (int i = 0; i < 4; ++i) {
            const _Float16* bp =
                &Bs[buf][(waveN * 64 + i * 16 + r) * KPAD + hiHalf * 8];
            v8h lo = *(const v8h*)bp;
            v8h hi = *(const v8h*)(bp + 16);
            v16h b;
#pragma unroll
            for (int j = 0; j < 8; ++j) { b[j] = lo[j]; b[j + 8] = hi[j]; }
            acc[0][i] = __builtin_amdgcn_wmma_f32_16x16x32_f16(
                false, a[0], false, b, (short)0, acc[0][i], false, false);
            acc[1][i] = __builtin_amdgcn_wmma_f32_16x16x32_f16(
                false, a[1], false, b, (short)0, acc[1][i], false, false);
        }
    };

    // ---- software pipeline: asyncB(k+1) | fetchA(k+1) | compute(k) ----
    fetchA(0);
    stageA(0);
    stageB_async(0, 0);
    asm volatile("s_wait_asynccnt 0x0" ::: "memory");
    __syncthreads();
    for (int kt = 0; kt < NKT; ++kt) {
        const int cur = kt & 1;
        if (kt + 1 < NKT) {
            stageB_async(cur ^ 1, kt + 1);   // DMA overlaps the WMMAs below
            fetchA(kt + 1);
        }
        compute(cur);
        if (kt + 1 < NKT) stageA(cur ^ 1);
        asm volatile("s_wait_asynccnt 0x0" ::: "memory");
        __syncthreads();
    }

    // ---- store: each lane owns 8 consecutive p for a fixed channel o.
    // Non-temporal: output is write-once; keep x/weights resident in L2.
    const size_t imgBase = (size_t)nImg * (OUT_CH * PIX);
#pragma unroll
    for (int i = 0; i < 2; ++i) {
        const int p0 = pBase + waveM * 32 + i * 16 + hiHalf * 8;
#pragma unroll
        for (int j = 0; j < 4; ++j) {
            const int o = waveN * 64 + j * 16 + r;
            const float bv = bias[o];
            float* dst = out + imgBase + (size_t)o * PIX + p0;
            v4f lo = { acc[i][j][0] + bv, acc[i][j][1] + bv,
                       acc[i][j][2] + bv, acc[i][j][3] + bv };
            v4f hi = { acc[i][j][4] + bv, acc[i][j][5] + bv,
                       acc[i][j][6] + bv, acc[i][j][7] + bv };
            __builtin_nontemporal_store(lo, (v4f*)dst);
            __builtin_nontemporal_store(hi, (v4f*)(dst + 4));
        }
    }
}

extern "C" void kernel_launch(void* const* d_in, const int* in_sizes, int n_in,
                              void* d_out, int out_size, void* d_ws, size_t ws_size,
                              hipStream_t stream) {
    (void)in_sizes; (void)n_in; (void)out_size; (void)ws_size;
    const float* x    = (const float*)d_in[0];
    const float* w    = (const float*)d_in[1];
    const float* bias = (const float*)d_in[2];
    float* out        = (float*)d_out;
    _Float16* bt      = (_Float16*)d_ws;   // 256*1152 f16 = 576 KB scratch

    reorder_weights_f16<<<(OUT_CH * KTOT) / 256, 256, 0, stream>>>(w, bt);

    dim3 grid((32 * PIX) / BM, 1);   // (1568, 1)
    conv3x3_wmma_f16<<<grid, 256, 0, stream>>>(x, bt, bias, out);
}
